// LIFbackEI_11879879544396
// MI455X (gfx1250) — matrix operand order repaired and checked by reading, not compiled, not run on hardware
//
#include <hip/hip_runtime.h>

// ---------------------------------------------------------------------------
// LIF-with-feedback-EI recurrence on MI455X (gfx1250, wave32, WMMA).
// Each time step = two 128->128 5x5 SAME convs as implicit GEMM on
// v_wmma_f32_16x16x32_f16 with fused LIF pointwise epilogues.
// Per workgroup: M=64 (2 output rows x 32 cols), N=128 (all co), K=3200.
// 8 waves: wave wv -> co tile wv*16, 4 accumulators (2 rows x 2 w-tiles).
// Software-pipelined fragment loads (1 iteration ahead), immediate-offset
// addressing for both LDS A fragments and L2-resident packed B fragments.
// ---------------------------------------------------------------------------

typedef __attribute__((ext_vector_type(16))) _Float16 v16h;
typedef __attribute__((ext_vector_type(8)))  _Float16 v8h;
typedef __attribute__((ext_vector_type(4)))  _Float16 v4h;
typedef __attribute__((ext_vector_type(8)))  float    v8f;

#define TT   8
#define NB   16
#define NC   128
#define NH   32
#define NW   32
#define KS   5
#define PADW 36           // NW + 2*2 halo columns
#define ROWS 6            // 2 output rows + 4 halo rows staged in LDS

// ---------------------------------------------------------------------------
// Pack f32 OIHW weights into f16 tiles laid out exactly as the WMMA B fragment:
//   pk[ kb ][ hf ][ co ][ k16 ],  kb = (kh*5+kw)*4 + ci/32, hf = (ci%32)/16,
//   k16 = ci%16.  Lane L then loads its whole v16h B fragment (K = 16*hf + e,
//   N = co) as one contiguous 32-byte read that stays L2-resident.
// ---------------------------------------------------------------------------
__global__ void pack_weights_kernel(const float* __restrict__ wback,
                                    const float* __restrict__ wei,
                                    _Float16* __restrict__ pback,
                                    _Float16* __restrict__ pei) {
    int idx = blockIdx.x * 256 + threadIdx.x;
    const int total = NC * NC * KS * KS;
    if (idx >= total) return;
    int co  = idx / (NC * KS * KS);
    int rem = idx % (NC * KS * KS);
    int ci  = rem / (KS * KS);
    int k   = rem % (KS * KS);          // kh*5 + kw
    int kb  = k * 4 + (ci >> 5);
    int c   = ci & 31;
    int hf  = c >> 4;
    int e   = c & 15;
    int dst = ((kb * 2 + hf) * NC + co) * 16 + e;
    pback[dst] = (_Float16)wback[idx];
    pei[dst]   = (_Float16)wei[idx];
}

// A-fragment LDS offset (halves) relative to per-lane base &lds[n*NC + hf*16]:
//   row = kb/20 + (tile>>1), x += kw + (tile&1)*16, chunk = (kb%4)*32
#define AOFF(kb, t) \
    (((((kb) / 20) + ((t) >> 1)) * PADW + (((kb) / 4) % 5) + (((t) & 1) * 16)) * NC \
     + ((kb) % 4) * 32)

// ---------------------------------------------------------------------------
template <bool EI>
__launch_bounds__(256)
__global__ void lif_conv_step(const _Float16* __restrict__ cin,   // [B][C][H][W] conv input (f16)
                              const _Float16* __restrict__ wpk,   // packed weights
                              const float*    __restrict__ bias,  // [C]
                              const float*    __restrict__ xt,    // x[t] (EI=false only)
                              float*          __restrict__ mem,   // [B][C][H][W] f32 state
                              _Float16*       __restrict__ aux16, // mem_f16 / spike_f16
                              float*          __restrict__ outp)  // spikes[t] (EI=true only)
{
    __shared__ _Float16 lds[ROWS * PADW * NC];   // 55,296 B: [row][x][ci]

    const int tid = threadIdx.x;
    const int b   = blockIdx.x >> 4;
    const int h0  = (blockIdx.x & 15) * 2;       // two output rows h0, h0+1

    // ---- stage 6 input rows (zero halo) into LDS --------------------------
    for (int r = 0; r < ROWS; ++r) {
        const int  hr   = h0 + r - 2;
        const bool vrow = (hr >= 0) && (hr < NH);
        // interior: 128 ci x 32 w, 4-half vector loads (coalesced in w)
        for (int it = 0; it < 4; ++it) {
            const int ci = tid & 127;
            const int wq = it * 2 + (tid >> 7);
            const int w  = wq << 2;
            v4h v = {};
            if (vrow)
                v = *(const v4h*)(cin + (((size_t)b * NC + ci) * NH + hr) * NW + w);
            _Float16* dst = &lds[(r * PADW + (w + 2)) * NC + ci];
#pragma unroll
            for (int j = 0; j < 4; ++j) dst[j * NC] = v[j];
        }
        // halo columns x = 0,1,34,35
        for (int it = 0; it < 2; ++it) {
            const int idx2 = it * 256 + tid;
            const int ci   = idx2 & 127;
            const int p    = idx2 >> 7;                 // 0..3
            const int x    = (p < 2) ? p : (32 + p);    // 0,1,34,35
            lds[(r * PADW + x) * NC + ci] = (_Float16)0.f;
        }
    }
    __syncthreads();

    // ---- WMMA implicit GEMM ----------------------------------------------
    const int lane = tid & 31;
    const int wave = tid >> 5;
    const int n    = lane & 15;     // N index (co) / A row (w) within tile
    const int hf   = lane >> 4;     // half-wave select
    const int co   = wave * 16 + n;

    // Per-lane bases; all loop offsets are compile-time immediates.
    const _Float16* bp = wpk + ((size_t)hf * NC + co) * 16;   // + kb*4096 halves
    const _Float16* ap = &lds[n * NC + hf * 16];              // + AOFF(kb,t)

    union U { v16h v; v8h h2[2]; };
    v8f acc[4] = {v8f{}, v8f{}, v8f{}, v8f{}};

    // prefetch kb = 0
    v16h bf = *(const v16h*)(bp);
    U a[4];
#pragma unroll
    for (int t = 0; t < 4; ++t) {
        a[t].h2[0] = *(const v8h*)(ap + AOFF(0, t));
        a[t].h2[1] = *(const v8h*)(ap + AOFF(0, t) + 8);
    }

#pragma unroll
    for (int kb = 0; kb < 100; ++kb) {
        v16h bn; U an[4];
        if (kb + 1 < 100) {                       // software pipeline: kb+1
            bn = *(const v16h*)(bp + (size_t)(kb + 1) * 4096);
#pragma unroll
            for (int t = 0; t < 4; ++t) {
                an[t].h2[0] = *(const v8h*)(ap + AOFF(kb + 1, t));
                an[t].h2[1] = *(const v8h*)(ap + AOFF(kb + 1, t) + 8);
            }
        }
#pragma unroll
        for (int t = 0; t < 4; ++t)
            acc[t] = __builtin_amdgcn_wmma_f32_16x16x32_f16(
                false, a[t].v, false, bf, (short)0, acc[t], false, false);
        if (kb + 1 < 100) {
            bf = bn;
#pragma unroll
            for (int t = 0; t < 4; ++t) a[t] = an[t];
        }
    }

    // ---- fused LIF epilogue ----------------------------------------------
    // C/D layout: lane L, VGPR e -> M = e + 8*hf (+16 per w-tile), N = L%16
    const float bia = bias[co];

#pragma unroll
    for (int mt = 0; mt < 4; ++mt) {
        const int    row = h0 + (mt >> 1);
        const int    w0  = (mt & 1) * 16 + hf * 8;
        const size_t g0  = (((size_t)b * NC + co) * NH + row) * NW + w0;
        const v8f    av  = acc[mt];
        if (!EI) {
#pragma unroll
            for (int e = 0; e < 8; ++e) {
                const size_t g = g0 + e;
                const float y    = av[e] + bia;
                const float gate = 1.f / (1.f + __expf(-y));
                const float mnew = 0.2f * mem[g] + gate * xt[g];
                mem[g]   = mnew;
                aux16[g] = (_Float16)mnew;
            }
        } else {
#pragma unroll
            for (int e = 0; e < 8; ++e) {
                const size_t g = g0 + e;
                const float z  = av[e] + bia;
                const float ei = (z > 0.f) ? 1.f : ((z < 0.f) ? -1.f : 0.f);
                const float mv = mem[g];
                const float s  = ((mv - 0.5f) > 0.f) ? 1.f : 0.f;
                mem[g]   = mv * (1.f - s);
                const float sp = ei * s;
                outp[g]  = sp;
                aux16[g] = (_Float16)sp;
            }
        }
    }
}

// ---------------------------------------------------------------------------
extern "C" void kernel_launch(void* const* d_in, const int* in_sizes, int n_in,
                              void* d_out, int out_size, void* d_ws, size_t ws_size,
                              hipStream_t stream) {
    (void)in_sizes; (void)n_in; (void)out_size; (void)ws_size;

    const float* x      = (const float*)d_in[0];   // [T,B,C,H,W]
    const float* back_w = (const float*)d_in[1];   // [C,C,5,5]
    const float* back_b = (const float*)d_in[2];   // [C]
    const float* ei_w   = (const float*)d_in[3];   // [C,C,5,5]
    const float* ei_b   = (const float*)d_in[4];   // [C]
    float* out = (float*)d_out;                    // [T,B,C,H,W]

    const size_t state = (size_t)NB * NC * NH * NW;    // 2,097,152 elements

    char* ws = (char*)d_ws;
    float*    mem   = (float*)(ws);                                // 8 MiB
    _Float16* memh  = (_Float16*)(ws + state * 4);                 // 4 MiB
    _Float16* spkh  = (_Float16*)(ws + state * 6);                 // 4 MiB
    _Float16* pback = (_Float16*)(ws + state * 8);                 // 800 KiB
    _Float16* pei   = pback + (size_t)NC * NC * KS * KS;           // 800 KiB

    // zero-init recurrent state (graph-capture-safe)
    hipMemsetAsync(mem,  0, state * sizeof(float),    stream);
    hipMemsetAsync(spkh, 0, state * sizeof(_Float16), stream);

    const int wtotal = NC * NC * KS * KS;
    pack_weights_kernel<<<(wtotal + 255) / 256, 256, 0, stream>>>(
        back_w, ei_w, pback, pei);

    const dim3 grid(NB * (NH / 2)), block(256);    // 256 workgroups
    for (int t = 0; t < TT; ++t) {
        const size_t off = (size_t)t * state;
        // mem = 0.2*mem + sigmoid(conv(spike, back_w)+b) * x[t]
        lif_conv_step<false><<<grid, block, 0, stream>>>(
            spkh, pback, back_b, x + off, mem, memh, nullptr);
        // spike = sign(conv(mem, ei_w)+b) * (mem > 0.5); mem reset
        lif_conv_step<true><<<grid, block, 0, stream>>>(
            memh, pei, ei_b, nullptr, mem, spkh, out + off);
    }
}
